// LLM_25786983645700
// MI455X (gfx1250) — matrix operand order repaired and checked by reading, compiled, not run on hardware
//
#include <hip/hip_runtime.h>
#include <hip/hip_bf16.h>
#include <math.h>

typedef __bf16 bf16;
typedef __attribute__((ext_vector_type(16))) __bf16 bf16x16;
typedef __attribute__((ext_vector_type(8)))  __bf16 bf16x8;
typedef __attribute__((ext_vector_type(8)))  float   f32x8;

// Model constants
#define CV   32000
#define CD   1024
#define CH   16
#define CHD  64
#define CL   8
#define CS   1024
#define CFF  4096

// Epilogue codes
#define EPI_F32            0   // plain f32 store (logits)
#define EPI_BIAS_RES_F32   1   // f32 = acc + bias[n] + resid[m,n]
#define EPI_BIAS_BF16      2   // bf16 = acc + bias[n]
#define EPI_BF16           3   // bf16 = acc
#define EPI_BIAS_GELU_BF16 4   // bf16 = gelu(acc + bias[n])
#define EPI_SCALE_CAUSAL   5   // f32 = (n<=m) ? acc*scale : -1e30
#define EPI_BIAS_BF16_T    6   // bf16 C^T store: C[n*ldc+m] = acc + bias[n]

// ---------------------------------------------------------------------------
// CDNA5 async global->LDS copy (ASYNCcnt-tracked), ISA §15.18 opcode 98
// ---------------------------------------------------------------------------
__device__ __forceinline__ void async_ld_b128(unsigned ldsOff, unsigned gOff,
                                              const void* base) {
    asm volatile("global_load_async_to_lds_b128 %0, %1, %2"
                 :: "v"(ldsOff), "v"(gOff), "s"(base)
                 : "memory");
}

template<int N>
__device__ __forceinline__ void wait_asynccnt() {
    asm volatile("s_wait_asynccnt %0" :: "n"(N) : "memory");
}

// ---------------------------------------------------------------------------
// Conversion kernels (run once per launch; weights -> bf16, B as [N,K])
// ---------------------------------------------------------------------------
__global__ __launch_bounds__(256)
void LLM_cvt_f32_bf16(const float* __restrict__ src, bf16* __restrict__ dst,
                      long long n) {
    long long i = (long long)blockIdx.x * 256 + threadIdx.x;
    if (i < n) dst[i] = (bf16)src[i];
}

// [L,H,D,HD] f32  ->  per-layer [N=H*HD, K=D] row-major bf16
__global__ __launch_bounds__(256)
void LLM_cvt_qkvT(const float* __restrict__ src, bf16* __restrict__ dst) {
    long long i = (long long)blockIdx.x * 256 + threadIdx.x;
    if (i >= (long long)CL * CH * CD * CHD) return;
    int l  = (int)(i >> 20);          // H*D*HD = 2^20
    int r  = (int)(i & 0xFFFFF);
    int h  = r >> 16;                 // D*HD = 2^16
    int r2 = r & 0xFFFF;
    int d  = r2 >> 6;                 // HD = 64
    int e  = r2 & 63;
    dst[(size_t)l * CD * CD + (size_t)(h * CHD + e) * CD + d] = (bf16)src[i];
}

// per-layer [K=R, N=C] f32 -> [N=C, K=R] bf16 (blockIdx.y = layer)
__global__ __launch_bounds__(256)
void LLM_cvt_T(const float* __restrict__ src, bf16* __restrict__ dst,
               int R, int C) {
    int i = blockIdx.x * 256 + threadIdx.x;
    if (i >= R * C) return;
    size_t base = (size_t)blockIdx.y * R * C;
    int k = i / C;
    int n = i - k * C;
    dst[base + (size_t)n * R + k] = (bf16)src[base + i];
}

// ---------------------------------------------------------------------------
// Embedding: x[s,d] = tok_emb[idx[s],d] + pos_emb[s,d]
// ---------------------------------------------------------------------------
__global__ __launch_bounds__(256)
void LLM_embed(const int* __restrict__ idx, const float* __restrict__ tok,
               const float* __restrict__ pos, float* __restrict__ x) {
    int i = blockIdx.x * 256 + threadIdx.x;
    if (i >= CS * CD) return;
    int s = i >> 10;
    int d = i & 1023;
    x[i] = tok[(size_t)idx[s] * CD + d] + pos[i];
}

// ---------------------------------------------------------------------------
// LayerNorm: f32 in -> bf16 out (GEMM A operand)
// ---------------------------------------------------------------------------
__global__ __launch_bounds__(256)
void LLM_ln(const float* __restrict__ x, const float* __restrict__ g,
            const float* __restrict__ b, bf16* __restrict__ out) {
    __shared__ float red[256];
    const int tid = threadIdx.x;
    const float* xr = x + (size_t)blockIdx.x * CD;
    float s = 0.f;
    for (int i = tid; i < CD; i += 256) s += xr[i];
    red[tid] = s; __syncthreads();
    for (int o = 128; o > 0; o >>= 1) {
        if (tid < o) red[tid] += red[tid + o];
        __syncthreads();
    }
    float mu = red[0] * (1.0f / CD);
    __syncthreads();
    float s2 = 0.f;
    for (int i = tid; i < CD; i += 256) { float d = xr[i] - mu; s2 += d * d; }
    red[tid] = s2; __syncthreads();
    for (int o = 128; o > 0; o >>= 1) {
        if (tid < o) red[tid] += red[tid + o];
        __syncthreads();
    }
    float inv = rsqrtf(red[0] * (1.0f / CD) + 1e-5f);
    bf16* orow = out + (size_t)blockIdx.x * CD;
    for (int i = tid; i < CD; i += 256)
        orow[i] = (bf16)((xr[i] - mu) * inv * g[i] + b[i]);
}

// ---------------------------------------------------------------------------
// Attention-row softmax: f32 scores [H*S, S] -> bf16 probabilities
// ---------------------------------------------------------------------------
__global__ __launch_bounds__(256)
void LLM_att_softmax(const float* __restrict__ sc, bf16* __restrict__ att) {
    __shared__ float red[256];
    const int tid = threadIdx.x;
    const float* p = sc + (size_t)blockIdx.x * CS;
    bf16* o = att + (size_t)blockIdx.x * CS;
    float m = -1e30f;
    for (int i = tid; i < CS; i += 256) m = fmaxf(m, p[i]);
    red[tid] = m; __syncthreads();
    for (int off = 128; off > 0; off >>= 1) {
        if (tid < off) red[tid] = fmaxf(red[tid], red[tid + off]);
        __syncthreads();
    }
    float mu = red[0];
    __syncthreads();
    float s = 0.f;
    for (int i = tid; i < CS; i += 256) s += expf(p[i] - mu);
    red[tid] = s; __syncthreads();
    for (int off = 128; off > 0; off >>= 1) {
        if (tid < off) red[tid] += red[tid + off];
        __syncthreads();
    }
    float inv = 1.0f / red[0];
    for (int i = tid; i < CS; i += 256)
        o[i] = (bf16)(expf(p[i] - mu) * inv);
}

// ---------------------------------------------------------------------------
// Vocab softmax in-place on d_out: [S, V] f32
// ---------------------------------------------------------------------------
__global__ __launch_bounds__(256)
void LLM_vocab_softmax(float* __restrict__ logits) {
    __shared__ float red[256];
    const int tid = threadIdx.x;
    float* p = logits + (size_t)blockIdx.x * CV;
    float m = -1e30f;
    for (int i = tid; i < CV; i += 256) m = fmaxf(m, p[i]);
    red[tid] = m; __syncthreads();
    for (int off = 128; off > 0; off >>= 1) {
        if (tid < off) red[tid] = fmaxf(red[tid], red[tid + off]);
        __syncthreads();
    }
    float mu = red[0];
    __syncthreads();
    float s = 0.f;
    for (int i = tid; i < CV; i += 256) s += expf(p[i] - mu);
    red[tid] = s; __syncthreads();
    for (int off = 128; off > 0; off >>= 1) {
        if (tid < off) red[tid] += red[tid + off];
        __syncthreads();
    }
    float inv = 1.0f / red[0];
    for (int i = tid; i < CV; i += 256)
        p[i] = expf(p[i] - mu) * inv;
}

// ---------------------------------------------------------------------------
// WMMA GEMM: C[bz][M,N] (+epi) = A[bz][M,K](bf16,[M,K]) x B[bz][N,K](bf16,[N,K])
//  Block: 256 threads = 8 wave32 -> 128x64 tile, BK=32, double-buffered LDS
//  filled by global_load_async_to_lds_b128 (ASYNCcnt), 2x2 WMMA per wave.
// ---------------------------------------------------------------------------
#define ASTR 40   // padded LDS row stride (bf16 elems)

template<int EPI>
__global__ __launch_bounds__(256)
void LLM_gemm(const bf16* __restrict__ A, int lda, long long strA,
              const bf16* __restrict__ B, int ldb, long long strB,
              void* __restrict__ Cp, int ldc, long long strC,
              int K,
              const float* __restrict__ bias,
              const float* __restrict__ resid,
              float scale) {
    __shared__ __align__(16) bf16 shA[2][128 * ASTR];
    __shared__ __align__(16) bf16 shB[2][64 * ASTR];

    const int tid  = threadIdx.x;
    const int bm   = blockIdx.y * 128;
    const int bn   = blockIdx.x * 64;
    const int bz   = blockIdx.z;
    const bf16* Ab = A + (long long)bz * strA;
    const bf16* Bb = B + (long long)bz * strB;

    const int lane = tid & 31;
    const int wave = tid >> 5;                 // 0..7, grid 4(m) x 2(n)
    const int wm   = (wave >> 1) * 32;
    const int wn   = (wave & 1) * 32;
    const int l15  = lane & 15;
    const int lkh  = lane >> 4;

    // issue one 128x32 A tile + 64x32 B tile as async 16B copies (3 per thread)
    auto issue_tile = [&](int buf, int kt) {
#pragma unroll
        for (int i = 0; i < 2; ++i) {
            int c   = tid + i * 256;           // 512 chunks of 8 bf16
            int row = c >> 2;                  // 0..127
            int kc  = (c & 3) << 3;
            unsigned lo = (unsigned)(size_t)&shA[buf][row * ASTR + kc];
            unsigned go = (unsigned)(((unsigned)(bm + row) * (unsigned)lda +
                                      (unsigned)(kt + kc)) * 2u);
            async_ld_b128(lo, go, Ab);
        }
        {
            int c  = tid;                      // 256 chunks
            int n  = c >> 2;                   // 0..63
            int kc = (c & 3) << 3;
            unsigned lo = (unsigned)(size_t)&shB[buf][n * ASTR + kc];
            unsigned go = (unsigned)(((unsigned)(bn + n) * (unsigned)ldb +
                                      (unsigned)(kt + kc)) * 2u);
            async_ld_b128(lo, go, Bb);
        }
    };

    f32x8 acc[2][2];
#pragma unroll
    for (int i = 0; i < 2; ++i)
#pragma unroll
        for (int j = 0; j < 2; ++j)
#pragma unroll
            for (int v = 0; v < 8; ++v) acc[i][j][v] = 0.0f;

    const int ntiles = K >> 5;
    issue_tile(0, 0);
    for (int t = 0; t < ntiles; ++t) {
        const int cur = t & 1;
        if (t + 1 < ntiles) {
            issue_tile(cur ^ 1, (t + 1) << 5);
            wait_asynccnt<3>();                // current tile landed
        } else {
            wait_asynccnt<0>();
        }
        __syncthreads();

        bf16x16 af[2], bff[2];
#pragma unroll
        for (int i = 0; i < 2; ++i) {
            const bf16* base = &shA[cur][(wm + i * 16 + l15) * ASTR];
            bf16x8 lo = *(const bf16x8*)(base + lkh * 8);       // K=kh*8+0..7
            bf16x8 hi = *(const bf16x8*)(base + 16 + lkh * 8);  // K=16+kh*8+0..7
#pragma unroll
            for (int e = 0; e < 8; ++e) { af[i][e] = lo[e]; af[i][8 + e] = hi[e]; }
        }
#pragma unroll
        for (int j = 0; j < 2; ++j) {
            const bf16* base = &shB[cur][(wn + j * 16 + l15) * ASTR];
            bf16x8 lo = *(const bf16x8*)(base + lkh * 8);       // K=kh*8+0..7
            bf16x8 hi = *(const bf16x8*)(base + 16 + lkh * 8);  // K=16+kh*8+0..7
#pragma unroll
            for (int e = 0; e < 8; ++e) { bff[j][e] = lo[e]; bff[j][8 + e] = hi[e]; }
        }
#pragma unroll
        for (int i = 0; i < 2; ++i)
#pragma unroll
            for (int j = 0; j < 2; ++j)
                acc[i][j] = __builtin_amdgcn_wmma_f32_16x16x32_bf16(
                    false, af[i], false, bff[j], (short)0, acc[i][j], false, false);
        __syncthreads();
    }

    // ---- epilogue: C/D layout M = v + 8*lkh, N = l15 ----
#pragma unroll
    for (int i = 0; i < 2; ++i)
#pragma unroll
        for (int j = 0; j < 2; ++j)
#pragma unroll
            for (int v = 0; v < 8; ++v) {
                int gm = bm + wm + i * 16 + lkh * 8 + v;
                int gn = bn + wn + j * 16 + l15;
                float val = acc[i][j][v];
                size_t cidx = (size_t)bz * (size_t)strC +
                              (size_t)gm * (size_t)ldc + gn;
                if (EPI == EPI_F32) {
                    ((float*)Cp)[cidx] = val;
                } else if (EPI == EPI_BIAS_RES_F32) {
                    ((float*)Cp)[cidx] = val + bias[gn] +
                                         resid[(size_t)gm * (size_t)ldc + gn];
                } else if (EPI == EPI_BIAS_BF16) {
                    ((bf16*)Cp)[cidx] = (bf16)(val + bias[gn]);
                } else if (EPI == EPI_BF16) {
                    ((bf16*)Cp)[cidx] = (bf16)val;
                } else if (EPI == EPI_BIAS_GELU_BF16) {
                    float t = val + bias[gn];
                    t = 0.5f * t * (1.0f + erff(t * 0.70710678118f));
                    ((bf16*)Cp)[cidx] = (bf16)t;
                } else if (EPI == EPI_SCALE_CAUSAL) {
                    float t = val * scale;
                    if (gn > gm) t = -1e30f;
                    ((float*)Cp)[cidx] = t;
                } else if (EPI == EPI_BIAS_BF16_T) {
                    size_t tix = (size_t)bz * (size_t)strC +
                                 (size_t)gn * (size_t)ldc + gm;
                    ((bf16*)Cp)[tix] = (bf16)(val + bias[gn]);
                }
            }
}

// ---------------------------------------------------------------------------
// Host launcher
// ---------------------------------------------------------------------------
extern "C" void kernel_launch(void* const* d_in, const int* in_sizes, int n_in,
                              void* d_out, int out_size, void* d_ws, size_t ws_size,
                              hipStream_t stream) {
    (void)in_sizes; (void)n_in; (void)out_size; (void)ws_size;

    const int*   idx   = (const int*)  d_in[0];
    const float* tok   = (const float*)d_in[1];
    const float* pos   = (const float*)d_in[2];
    const float* Wq    = (const float*)d_in[3];
    const float* bq    = (const float*)d_in[4];
    const float* Wk    = (const float*)d_in[5];
    const float* bk    = (const float*)d_in[6];
    const float* Wv    = (const float*)d_in[7];
    const float* bv    = (const float*)d_in[8];
    const float* Wp    = (const float*)d_in[9];
    const float* bp    = (const float*)d_in[10];
    const float* W1    = (const float*)d_in[11];
    const float* b1    = (const float*)d_in[12];
    const float* W2    = (const float*)d_in[13];
    const float* b2    = (const float*)d_in[14];
    const float* ln1g  = (const float*)d_in[15];
    const float* ln1b  = (const float*)d_in[16];
    const float* ln2g  = (const float*)d_in[17];
    const float* ln2b  = (const float*)d_in[18];
    const float* lnfg  = (const float*)d_in[19];
    const float* lnfb  = (const float*)d_in[20];

    // ---- workspace carve-out (256B aligned) ----
    char* p = (char*)d_ws;
    auto alloc = [&](size_t bytes) -> char* {
        char* r = p;
        p += (bytes + 255) & ~(size_t)255;
        return r;
    };
    const size_t nQKV = (size_t)CL * CH * CD * CHD;    // 8M per tensor
    bf16*  wqb   = (bf16*) alloc(nQKV * 2);            // [N=D, K=D] per layer
    bf16*  wkb   = (bf16*) alloc(nQKV * 2);
    bf16*  wvb   = (bf16*) alloc(nQKV * 2);
    bf16*  wpb   = (bf16*) alloc((size_t)CL * CD * CD * 2);   // [N=D, K=D]
    bf16*  w1b   = (bf16*) alloc((size_t)CL * CD * CFF * 2);  // [N=FF, K=D]
    bf16*  w2b   = (bf16*) alloc((size_t)CL * CFF * CD * 2);  // [N=D, K=FF]
    bf16*  tokb  = (bf16*) alloc((size_t)CV * CD * 2);        // [N=V, K=D]
    float* x     = (float*)alloc((size_t)CS * CD * 4);
    bf16*  h     = (bf16*) alloc((size_t)CS * CD * 2);
    bf16*  qb    = (bf16*) alloc((size_t)CS * CD * 2);        // [S, H*HD]
    bf16*  kb    = (bf16*) alloc((size_t)CS * CD * 2);        // [S, H*HD]
    bf16*  vtb   = (bf16*) alloc((size_t)CD * CS * 2);        // [H*HD, S]
    float* sc    = (float*)alloc((size_t)CH * CS * CS * 4);
    bf16*  att   = (bf16*) alloc((size_t)CH * CS * CS * 2);
    bf16*  ob    = (bf16*) alloc((size_t)CS * CD * 2);
    bf16*  h2    = (bf16*) alloc((size_t)CS * CD * 2);
    bf16*  gbuf  = (bf16*) alloc((size_t)CS * CFF * 2);
    bf16*  hf    = (bf16*) alloc((size_t)CS * CD * 2);

    // ---- weight conversions (bf16, all B operands as [N,K]) ----
    {
        int g = (int)((nQKV + 255) / 256);
        LLM_cvt_qkvT<<<g, 256, 0, stream>>>(Wq, wqb);
        LLM_cvt_qkvT<<<g, 256, 0, stream>>>(Wk, wkb);
        LLM_cvt_qkvT<<<g, 256, 0, stream>>>(Wv, wvb);
        LLM_cvt_T<<<dim3((CD * CD) / 256, CL), 256, 0, stream>>>(Wp, wpb, CD, CD);
        LLM_cvt_T<<<dim3((CD * CFF) / 256, CL), 256, 0, stream>>>(W1, w1b, CD, CFF);
        LLM_cvt_T<<<dim3((CFF * CD) / 256, CL), 256, 0, stream>>>(W2, w2b, CFF, CD);
        long long nT = (long long)CV * CD;
        LLM_cvt_f32_bf16<<<(int)((nT + 255) / 256), 256, 0, stream>>>(tok, tokb, nT);
    }

    // ---- embedding ----
    LLM_embed<<<(CS * CD) / 256, 256, 0, stream>>>(idx, tok, pos, x);

    const dim3 blk(256);
    const float attScale = 0.03125f;   // 1/sqrt(D) = 1/32

    for (int l = 0; l < CL; ++l) {
        const bf16* wql = wqb + (size_t)l * CD * CD;
        const bf16* wkl = wkb + (size_t)l * CD * CD;
        const bf16* wvl = wvb + (size_t)l * CD * CD;
        const bf16* wpl = wpb + (size_t)l * CD * CD;
        const bf16* w1l = w1b + (size_t)l * CD * CFF;
        const bf16* w2l = w2b + (size_t)l * CFF * CD;

        // LN1 -> h (bf16)
        LLM_ln<<<CS, 256, 0, stream>>>(x, ln1g + (size_t)l * CD,
                                       ln1b + (size_t)l * CD, h);
        // Q,K : [S,D]x[D,D]^T + bias -> bf16 [S, H*HD]
        dim3 gQKV(CD / 64, CS / 128, 1);
        LLM_gemm<EPI_BIAS_BF16><<<gQKV, blk, 0, stream>>>(
            h, CD, 0, wql, CD, 0, qb, CD, 0, CD, bq + (size_t)l * CD, nullptr, 1.f);
        LLM_gemm<EPI_BIAS_BF16><<<gQKV, blk, 0, stream>>>(
            h, CD, 0, wkl, CD, 0, kb, CD, 0, CD, bk + (size_t)l * CD, nullptr, 1.f);
        // V stored transposed: vtb[h*HD+e][s]
        LLM_gemm<EPI_BIAS_BF16_T><<<gQKV, blk, 0, stream>>>(
            h, CD, 0, wvl, CD, 0, vtb, CS, 0, CD, bv + (size_t)l * CD, nullptr, 1.f);

        // scores[h] = scale * q_h k_h^T with causal mask (batched over heads)
        dim3 gSc(CS / 64, CS / 128, CH);
        LLM_gemm<EPI_SCALE_CAUSAL><<<gSc, blk, 0, stream>>>(
            qb, CD, CHD, kb, CD, CHD, sc, CS, (long long)CS * CS,
            CHD, nullptr, nullptr, attScale);

        // softmax over keys -> bf16 att
        LLM_att_softmax<<<CH * CS, 256, 0, stream>>>(sc, att);

        // o[h] = att_h x v_h (B = vtb [N=HD, K=S]) -> bf16 [S, H*HD]
        dim3 gAV(CHD / 64, CS / 128, CH);
        LLM_gemm<EPI_BF16><<<gAV, blk, 0, stream>>>(
            att, CS, (long long)CS * CS, vtb, CS, (long long)CHD * CS,
            ob, CD, CHD, CS, nullptr, nullptr, 1.f);

        // x = x + o Wp + bp
        LLM_gemm<EPI_BIAS_RES_F32><<<gQKV, blk, 0, stream>>>(
            ob, CD, 0, wpl, CD, 0, x, CD, 0, CD, bp + (size_t)l * CD, x, 1.f);

        // LN2 -> h2
        LLM_ln<<<CS, 256, 0, stream>>>(x, ln2g + (size_t)l * CD,
                                       ln2b + (size_t)l * CD, h2);
        // gbuf = gelu(h2 W1 + b1)  (bf16)
        dim3 gM1(CFF / 64, CS / 128, 1);
        LLM_gemm<EPI_BIAS_GELU_BF16><<<gM1, blk, 0, stream>>>(
            h2, CD, 0, w1l, CD, 0, gbuf, CFF, 0, CD,
            b1 + (size_t)l * CFF, nullptr, 1.f);
        // x = x + gbuf W2 + b2
        LLM_gemm<EPI_BIAS_RES_F32><<<gQKV, blk, 0, stream>>>(
            gbuf, CFF, 0, w2l, CFF, 0, x, CD, 0, CFF,
            b2 + (size_t)l * CD, x, 1.f);
    }

    // final LN -> hf
    LLM_ln<<<CS, 256, 0, stream>>>(x, lnfg, lnfb, hf);

    // logits = hf tok_emb^T  (tokb is [V,D] = [N,K]), f32 into d_out
    dim3 gLg(CV / 64, CS / 128, 1);
    LLM_gemm<EPI_F32><<<gLg, blk, 0, stream>>>(
        hf, CD, 0, tokb, CD, 0, (float*)d_out, CV, 0, CD,
        nullptr, nullptr, 1.f);

    // softmax over vocab, in-place on d_out
    LLM_vocab_softmax<<<CS, 256, 0, stream>>>((float*)d_out);
}
